// StandardAttention_88562225643973
// MI455X (gfx1250) — compile-verified, matrix operand order checked by reading, not run on hardware
//
#include <hip/hip_runtime.h>

typedef __attribute__((ext_vector_type(16))) _Float16 v16h;
typedef __attribute__((ext_vector_type(8)))  float    v8f;
typedef __attribute__((ext_vector_type(4)))  float    v4f;

#define B_ 4
#define T_ 1024
#define S_ 1024
#define H_ 16
#define D_ 64
#define TM 16            // T rows per workgroup
#define NW 8             // waves per workgroup
#define SCHUNK (S_ / NW) // 128 S-columns per wave

__global__ __launch_bounds__(256)
void attn_fwd_wmma_kernel(const float* __restrict__ q,
                          const float* __restrict__ k,
                          const float* __restrict__ v,
                          float* __restrict__ attn,
                          float* __restrict__ probs)
{
    __shared__ float lds_p[TM * S_];     // 64 KB: logits -> exp -> probs
    __shared__ float lds_attn[TM * D_];  // 4 KB: cross-wave PV accumulator
    __shared__ float lds_red[TM * 16];   // softmax partials
    __shared__ float lds_max[TM];
    __shared__ float lds_inv[TM];

    const int tid  = threadIdx.x;
    const int wave = tid >> 5;
    const int lane = tid & 31;
    const int l16  = lane & 15;
    const int hi   = lane >> 4;          // lane-half (K/base-8 select)

    const int tTiles = T_ / TM;
    const int t0 = (blockIdx.x % tTiles) * TM;
    const int h  = (blockIdx.x / tTiles) % H_;
    const int b  =  blockIdx.x / (tTiles * H_);

    // zero the attn accumulator
    for (int i = tid; i < TM * D_; i += 256) lds_attn[i] = 0.0f;

    const float qscale = 0.125f;  // (D^-1/4) on both q,k folded == 1/8 on q

    // ---- Q A-fragments: 16x64 as two 16x32 f16 fragments -----------------
    // 16-bit A layout: lanes 0-15 row M=l16 hold K={0..7,16..23}; lanes 16-31
    // hold K={8..15,24..31}. element e -> K = e + (e>=8 ? 8 : 0) + hi*8.
    v16h a0, a1;
    {
        const float* qrow = q + (((size_t)b * T_ + (t0 + l16)) * H_ + h) * D_;
#pragma unroll
        for (int e = 0; e < 16; ++e) {
            int kk = e + ((e >= 8) ? 8 : 0) + hi * 8;
            a0[e] = (_Float16)(qrow[kk]      * qscale);
            a1[e] = (_Float16)(qrow[kk + 32] * qscale);
        }
    }

    // ---- QK^T for this wave's 128-column chunk ---------------------------
    const int sBase = wave * SCHUNK;
    v8f c[8];
#pragma unroll
    for (int j = 0; j < 8; ++j) {
        const int s0 = sBase + j * 16;
        // B (32x16): lane n holds column s=s0+n, K(d) per same half-split.
        const float* krow = k + (((size_t)b * S_ + (s0 + l16)) * H_ + h) * D_;
        v16h b0, b1;
#pragma unroll
        for (int e = 0; e < 16; ++e) {
            int kk = e + ((e >= 8) ? 8 : 0) + hi * 8;
            b0[e] = (_Float16)krow[kk];
            b1[e] = (_Float16)krow[kk + 32];
        }
        v8f acc = {};
        acc = __builtin_amdgcn_wmma_f32_16x16x32_f16(false, a0, false, b0,
                                                     (short)0, acc, false, false);
        acc = __builtin_amdgcn_wmma_f32_16x16x32_f16(false, a1, false, b1,
                                                     (short)0, acc, false, false);
        c[j] = acc;
    }

    // ---- spill logits to LDS (C/D layout: vgpr r -> M=r+hi*8, N=l16) -----
#pragma unroll
    for (int j = 0; j < 8; ++j) {
        const int s = sBase + j * 16 + l16;
#pragma unroll
        for (int r = 0; r < 8; ++r) {
            const int m = r + hi * 8;
            lds_p[m * S_ + s] = c[j][r];
        }
    }
    __syncthreads();

    // ---- softmax: row max, then exp in place + row sum --------------------
    {
        const int row = tid >> 4;   // 0..15
        const int sub = tid & 15;   // 16 threads/row, stride 16 (bank-clean)
        float mx = -3.4e38f;
        for (int s = sub; s < S_; s += 16) mx = fmaxf(mx, lds_p[row * S_ + s]);
        lds_red[row * 16 + sub] = mx;
        __syncthreads();
        if (tid < TM) {
            float m2 = lds_red[tid * 16];
            for (int i = 1; i < 16; ++i) m2 = fmaxf(m2, lds_red[tid * 16 + i]);
            lds_max[tid] = m2;
        }
        __syncthreads();
        const float rowmax = lds_max[row];
        float sum = 0.0f;
        for (int s = sub; s < S_; s += 16) {
            float e = __expf(lds_p[row * S_ + s] - rowmax);
            lds_p[row * S_ + s] = e;     // exp stored back; normalize later
            sum += e;
        }
        lds_red[row * 16 + sub] = sum;
        __syncthreads();
        if (tid < TM) {
            float s2 = 0.0f;
            for (int i = 0; i < 16; ++i) s2 += lds_red[tid * 16 + i];
            lds_inv[tid] = 1.0f / s2;
        }
        __syncthreads();
    }

    // ---- normalize + 128-bit non-temporal probs store ---------------------
    // Each iteration: 256 threads x float4 = one contiguous 4KB row.
    {
        v4f* ldsv = (v4f*)lds_p;
        for (int idx4 = tid; idx4 < (TM * S_) / 4; idx4 += 256) {
            const int m = idx4 >> 8;             // 256 float4 per row
            const int s4 = idx4 & 255;
            const float inv = lds_inv[m];
            v4f p = ldsv[idx4];
            p.x *= inv; p.y *= inv; p.z *= inv; p.w *= inv;
            ldsv[idx4] = p;                      // keep normalized probs for PV
            v4f* prow = (v4f*)(probs + (((size_t)b * T_ + (t0 + m)) * H_ + h) * S_);
            __builtin_nontemporal_store(p, prow + s4);  // streamed, bypass cache
        }
    }
    __syncthreads();

    // ---- P x V over this wave's chunk: (16x128) x (128x64) ---------------
    {
        v8f o0 = {}, o1 = {}, o2 = {}, o3 = {};
#pragma unroll
        for (int kk = 0; kk < 4; ++kk) {
            const int sb = sBase + kk * 32;
            // A-fragment from LDS: two contiguous 8-col groups -> 4x b128
            v4f g0 = *(const v4f*)&lds_p[l16 * S_ + sb + hi * 8];
            v4f g1 = *(const v4f*)&lds_p[l16 * S_ + sb + hi * 8 + 4];
            v4f g2 = *(const v4f*)&lds_p[l16 * S_ + sb + 16 + hi * 8];
            v4f g3 = *(const v4f*)&lds_p[l16 * S_ + sb + 16 + hi * 8 + 4];
            v16h ap;
            ap[0]  = (_Float16)g0.x; ap[1]  = (_Float16)g0.y;
            ap[2]  = (_Float16)g0.z; ap[3]  = (_Float16)g0.w;
            ap[4]  = (_Float16)g1.x; ap[5]  = (_Float16)g1.y;
            ap[6]  = (_Float16)g1.z; ap[7]  = (_Float16)g1.w;
            ap[8]  = (_Float16)g2.x; ap[9]  = (_Float16)g2.y;
            ap[10] = (_Float16)g2.z; ap[11] = (_Float16)g2.w;
            ap[12] = (_Float16)g3.x; ap[13] = (_Float16)g3.y;
            ap[14] = (_Float16)g3.z; ap[15] = (_Float16)g3.w;

            const float* vbase = v + (((size_t)b * S_) * H_ + h) * D_;
#pragma unroll
            for (int dt = 0; dt < 4; ++dt) {
                const float* vcol = vbase + dt * 16 + l16;
                v16h bp;
#pragma unroll
                for (int e = 0; e < 16; ++e) {
                    int so = e + ((e >= 8) ? 8 : 0) + hi * 8;
                    bp[e] = (_Float16)vcol[(size_t)(sb + so) * (H_ * D_)];
                }
                v8f* o = (dt == 0) ? &o0 : (dt == 1) ? &o1 : (dt == 2) ? &o2 : &o3;
                *o = __builtin_amdgcn_wmma_f32_16x16x32_f16(false, ap, false, bp,
                                                            (short)0, *o, false, false);
            }
        }
        // cross-wave reduction via LDS float atomics (ds_add_f32)
        v8f oo[4] = {o0, o1, o2, o3};
#pragma unroll
        for (int dt = 0; dt < 4; ++dt) {
#pragma unroll
            for (int r = 0; r < 8; ++r) {
                const int m = r + hi * 8;
                const int d = dt * 16 + l16;
                atomicAdd(&lds_attn[m * D_ + d], oo[dt][r]);
            }
        }
    }
    __syncthreads();

    // ---- 128-bit coalesced attn store (one float4 per thread) ------------
    {
        const int idx4 = tid;                    // 256 float4 == 16x64 floats
        const int m  = idx4 >> 4;                // 16 float4 per row
        const int d4 = idx4 & 15;
        v4f val = ((v4f*)lds_attn)[idx4];
        v4f* arow = (v4f*)(attn + (((size_t)b * T_ + (t0 + m)) * H_ + h) * D_);
        arow[d4] = val;
    }
}

extern "C" void kernel_launch(void* const* d_in, const int* in_sizes, int n_in,
                              void* d_out, int out_size, void* d_ws, size_t ws_size,
                              hipStream_t stream)
{
    (void)in_sizes; (void)n_in; (void)d_ws; (void)ws_size; (void)out_size;
    const float* q = (const float*)d_in[0];
    const float* k = (const float*)d_in[1];
    const float* v = (const float*)d_in[2];
    float* attn  = (float*)d_out;                              // (B,T,H,D)
    float* probs = attn + (size_t)B_ * T_ * H_ * D_;           // (B,T,H,S)

    const int blocks = B_ * H_ * (T_ / TM);  // 4096
    attn_fwd_wmma_kernel<<<blocks, 256, 0, stream>>>(q, k, v, attn, probs);
}
